// TopKSAE_68324339745163
// MI455X (gfx1250) — matrix operand order repaired and checked by reading, not compiled, not run on hardware
//
#include <hip/hip_runtime.h>

#define D_IN  768
#define NLAT  24576
#define TOPK  64
#define KTILE 32
#define BM    128
#define BN    256
#define NTW   (BN / 16)          // 16 N sub-tiles per wave
#define LROW  40                 // LDS row stride in ushorts (80B, 16B aligned)

typedef __attribute__((ext_vector_type(16))) __bf16        v16bf;
typedef __attribute__((ext_vector_type(8)))  float         v8f;
typedef __attribute__((ext_vector_type(4)))  float         v4f;
typedef __attribute__((ext_vector_type(4)))  unsigned int  v4u;

union Frag { v4u q[2]; v16bf b; };

__device__ __forceinline__ unsigned short f2bf(float f) {
    unsigned u = __float_as_uint(f);
    unsigned r = u + 0x7FFFu + ((u >> 16) & 1u);   // round-to-nearest-even
    return (unsigned short)(r >> 16);
}

__device__ __forceinline__ unsigned f2key(float f) {
    unsigned u = __float_as_uint(f);
    return ((int)u < 0) ? ~u : (u | 0x80000000u);  // monotonic total order
}

__device__ __forceinline__ float key2f(unsigned k) {
    unsigned u = (k & 0x80000000u) ? (k & 0x7FFFFFFFu) : ~k;
    return __uint_as_float(u);
}

// ---------------------------------------------------------------------------
// z_pre[M,N] = (x - b_pre)[M,K] @ W_enc[N,K]^T : bf16 WMMA, f32 accumulate,
// 128x256 block tile, double-buffered LDS panels, pipelined B fragments.
// ---------------------------------------------------------------------------
__global__ __launch_bounds__(256)
void sae_encode_gemm(const float* __restrict__ x,
                     const float* __restrict__ Wenc,
                     const float* __restrict__ bpre,
                     float* __restrict__ zpre)
{
    __shared__ __align__(16) unsigned short As[2][BM * LROW];
    __shared__ __align__(16) unsigned short Bs[2][BN * LROW];

    const int tid  = threadIdx.x;
    const int lane = tid & 31;
    const int ln15 = lane & 15;
    const int wave = tid >> 5;                 // 0..7 -> M sub-tile
    const int bm   = blockIdx.y * BM;
    const int bn   = blockIdx.x * BN;
    const int aq0  = (lane < 16) ? 0 : 1;      // A chunk select (16B units)
    const int bq0  = (lane < 16) ? 0 : 1;      // B chunk select (16B units)

    v8f acc[NTW];
#pragma unroll
    for (int t = 0; t < NTW; ++t)
#pragma unroll
        for (int j = 0; j < 8; ++j) acc[t][j] = 0.0f;

    auto stage = [&](int kt, int buf) {
        // A panel: 128 x 32 f32 -> bf16 (bias-subtracted); 1024 quads
#pragma unroll
        for (int i = 0; i < 4; ++i) {
            int lin = i * 256 + tid;
            int row = lin >> 3;
            int kq  = (lin & 7) * 4;
            v4f a = *(const v4f*)(x + (size_t)(bm + row) * D_IN + kt + kq);
            float b0 = bpre[kt + kq + 0], b1 = bpre[kt + kq + 1];
            float b2 = bpre[kt + kq + 2], b3 = bpre[kt + kq + 3];
            unsigned short* da = &As[buf][row * LROW + kq];
            da[0] = f2bf(a.x - b0); da[1] = f2bf(a.y - b1);
            da[2] = f2bf(a.z - b2); da[3] = f2bf(a.w - b3);
        }
        // B panel: 256 x 32 f32 -> bf16; 2048 quads
#pragma unroll
        for (int i = 0; i < 8; ++i) {
            int lin = i * 256 + tid;
            int row = lin >> 3;
            int kq  = (lin & 7) * 4;
            v4f w = *(const v4f*)(Wenc + (size_t)(bn + row) * D_IN + kt + kq);
            unsigned short* db = &Bs[buf][row * LROW + kq];
            db[0] = f2bf(w.x); db[1] = f2bf(w.y);
            db[2] = f2bf(w.z); db[3] = f2bf(w.w);
        }
    };

    auto compute = [&](int buf) {
        // A fragment: lane L -> row wave*16 + L%16;
        // lanes 0-15: K 0-7 & 16-23, lanes 16-31: K 8-15 & 24-31
        Frag af;
        {
            const v4u* arow = (const v4u*)&As[buf][(wave * 16 + ln15) * LROW];
            af.q[0] = arow[aq0];
            af.q[1] = arow[aq0 + 2];
        }
        // B fragments: lane L -> col L%16; lanes 0-15 K 0-7 & 16-23, lanes 16-31
        // K 8-15 & 24-31. Software-pipelined: load t+1 before WMMA t
        // so the dscnt wait overlaps LDS latency with the matrix op.
        auto loadB = [&](int t) {
            const v4u* brow = (const v4u*)&Bs[buf][(t * 16 + ln15) * LROW];
            Frag bf;
            bf.q[0] = brow[bq0];
            bf.q[1] = brow[bq0 + 2];
            return bf;
        };
        Frag bcur = loadB(0);
#pragma unroll
        for (int t = 0; t < NTW; ++t) {
            Frag bnext = bcur;
            if (t + 1 < NTW) bnext = loadB(t + 1);
            acc[t] = __builtin_amdgcn_wmma_f32_16x16x32_bf16(
                false, af.b, false, bcur.b, (short)0, acc[t], false, false);
            bcur = bnext;
        }
    };

    const int nk = D_IN / KTILE;               // 24 K-steps
    stage(0, 0);
    __syncthreads();
    int cur = 0;
    for (int ks = 0; ks < nk; ++ks) {
        if (ks + 1 < nk) stage((ks + 1) * KTILE, cur ^ 1);
        compute(cur);
        __syncthreads();
        cur ^= 1;
    }

    // D layout: VGPR j -> M = j (lanes 0-15) / j+8 (lanes 16-31); N = lane%16.
    // Streaming output -> non-temporal stores (keep W_enc L2-resident).
    const int mbase = bm + wave * 16 + ((lane < 16) ? 0 : 8);
#pragma unroll
    for (int t = 0; t < NTW; ++t) {
        const int n = bn + t * 16 + ln15;
#pragma unroll
        for (int j = 0; j < 8; ++j)
            __builtin_nontemporal_store(acc[t][j],
                &zpre[(size_t)(mbase + j) * NLAT + n]);
    }
}

// ---------------------------------------------------------------------------
// Exact top-64 per row. The whole row's sort keys live in LDS (96KB of the
// 320KB/WGP), so z_pre is streamed from HBM exactly once. 4 MSB-first radix
// passes find the 64th-largest key; deterministic bitmask+scan compaction
// emits (idx,val) pairs; dense z written with NT stores.
// ---------------------------------------------------------------------------
__global__ __launch_bounds__(256)
void sae_topk(const float* __restrict__ zpre,
              float* __restrict__ z,
              int*   __restrict__ ws_idx,
              float* __restrict__ ws_val)
{
    __shared__ __align__(16) unsigned keys[NLAT];   // 96 KB
    __shared__ unsigned hist[256];
    __shared__ unsigned scnt[256];
    __shared__ unsigned sh_prefix, sh_kneed, sh_tie;

    const int row = blockIdx.x;
    const int tid = threadIdx.x;
    const float* zp = zpre + (size_t)row * NLAT;

    if (tid == 0) { sh_prefix = 0u; sh_kneed = TOPK; sh_tie = 0u; }
    if (tid < TOPK) { ws_idx[row * TOPK + tid] = 0; ws_val[row * TOPK + tid] = 0.0f; }

    // single streaming read of the row; convert to sortable keys in LDS
    for (int base = tid * 4; base < NLAT; base += 1024) {
        v4f v = __builtin_nontemporal_load((const v4f*)(zp + base));
        v4u kk;
#pragma unroll
        for (int c = 0; c < 4; ++c) kk[c] = f2key(v[c]);
        *(v4u*)&keys[base] = kk;
    }
    __syncthreads();

    // 4 radix passes (8 bits, MSB first) -> exact 64th-largest key
    for (int pass = 0; pass < 4; ++pass) {
        hist[tid] = 0u;
        __syncthreads();
        const unsigned pre  = sh_prefix;
        const int     shift = 24 - 8 * pass;
        for (int i = tid; i < NLAT; i += 256) {
            unsigned key = keys[i];
            bool match = (pass == 0) || ((key >> (shift + 8)) == pre);
            if (match) atomicAdd(&hist[(key >> shift) & 255u], 1u);
        }
        __syncthreads();
        if (tid == 0) {
            unsigned need = sh_kneed, accum = 0u;
            int b = 255;
            for (; b > 0; --b) {
                unsigned nc = accum + hist[b];
                if (nc >= need) break;
                accum = nc;
            }
            sh_prefix = (pre << 8) | (unsigned)b;
            sh_kneed  = need - accum;
        }
        __syncthreads();
    }

    const unsigned T = sh_prefix;   // exact key of the 64th-largest element
    const unsigned r = sh_kneed;    // ties at T to keep

    // selection: write dense z (NT), record per-thread 96-bit selection mask
    unsigned mask[3] = {0u, 0u, 0u};
    unsigned cnt = 0u;
    int j = 0;
    for (int i = tid; i < NLAT; i += 256, ++j) {
        unsigned key = keys[i];
        bool sel = (key > T);
        if (!sel && key == T) sel = (atomicAdd(&sh_tie, 1u) < r);
        float v = sel ? key2f(key) : 0.0f;
        __builtin_nontemporal_store(v, &z[(size_t)row * NLAT + i]);
        if (sel) { mask[j >> 5] |= (1u << (j & 31)); ++cnt; }
    }
    scnt[tid] = cnt;
    __syncthreads();
    if (tid == 0) {
        unsigned run = 0u;
        for (int t = 0; t < 256; ++t) { unsigned c = scnt[t]; scnt[t] = run; run += c; }
    }
    __syncthreads();

    unsigned slot = scnt[tid];
    j = 0;
    for (int i = tid; i < NLAT; i += 256, ++j) {
        if ((mask[j >> 5] >> (j & 31)) & 1u) {
            if (slot < TOPK) {
                ws_idx[row * TOPK + slot] = i;
                ws_val[row * TOPK + slot] = key2f(keys[i]);
            }
            ++slot;
        }
    }
}

// ---------------------------------------------------------------------------
// Sparse decode: x_hat[row] = sum_j val[j] * W_dec[idx[j]] + b_pre;
// recon_error = x - x_hat. W_dec (75MB) stays L2-resident across blocks.
// ---------------------------------------------------------------------------
__global__ __launch_bounds__(256)
void sae_decode(const float* __restrict__ x,
                const float* __restrict__ Wdec,
                const float* __restrict__ bpre,
                const int*   __restrict__ ws_idx,
                const float* __restrict__ ws_val,
                float* __restrict__ xhat,
                float* __restrict__ rerr)
{
    __shared__ int   sidx[TOPK];
    __shared__ float sval[TOPK];
    const int row = blockIdx.x;
    const int tid = threadIdx.x;
    if (tid < TOPK) { sidx[tid] = ws_idx[row * TOPK + tid]; sval[tid] = ws_val[row * TOPK + tid]; }
    __syncthreads();

#pragma unroll
    for (int c = 0; c < D_IN / 256; ++c) {
        const int col = tid + c * 256;
        float acc = bpre[col];
#pragma unroll 8
        for (int jj = 0; jj < TOPK; ++jj)
            acc = fmaf(sval[jj], Wdec[(size_t)sidx[jj] * D_IN + col], acc);
        const size_t o = (size_t)row * D_IN + col;
        __builtin_nontemporal_store(acc, &xhat[o]);
        __builtin_nontemporal_store(x[o] - acc, &rerr[o]);
    }
}

// ---------------------------------------------------------------------------
extern "C" void kernel_launch(void* const* d_in, const int* in_sizes, int n_in,
                              void* d_out, int out_size, void* d_ws, size_t ws_size,
                              hipStream_t stream)
{
    const float* x    = (const float*)d_in[0];
    const float* Wenc = (const float*)d_in[1];
    const float* Wdec = (const float*)d_in[2];
    const float* bpre = (const float*)d_in[3];
    // d_in[4] = k (device scalar) — reference uses k = 64, baked in as TOPK.

    const int Batch = in_sizes[0] / D_IN;      // 4096

    float* out  = (float*)d_out;               // [x_hat | z | z_pre | recon]
    float* xhat = out;
    float* z    = xhat + (size_t)Batch * D_IN;
    float* zpre = z    + (size_t)Batch * NLAT;
    float* rerr = zpre + (size_t)Batch * NLAT;

    int*   ws_idx = (int*)d_ws;
    float* ws_val = (float*)((char*)d_ws + (size_t)Batch * TOPK * sizeof(int));

    dim3 gGemm(NLAT / BN, Batch / BM);
    sae_encode_gemm<<<gGemm, 256, 0, stream>>>(x, Wenc, bpre, zpre);
    sae_topk      <<<Batch, 256, 0, stream>>>(zpre, z, ws_idx, ws_val);
    sae_decode    <<<Batch, 256, 0, stream>>>(x, Wdec, bpre, ws_idx, ws_val, xhat, rerr);
}
